// SmVAE_16312285790242
// MI455X (gfx1250) — matrix operand order repaired and checked by reading, not compiled, not run on hardware
//
#include <hip/hip_runtime.h>
#include <hip/hip_bf16.h>

typedef __bf16 bhalf;
typedef __attribute__((ext_vector_type(16))) __bf16 v16bf;
typedef __attribute__((ext_vector_type(8)))  __bf16 v8bf;
typedef __attribute__((ext_vector_type(8)))  float  v8f;

#define VOCABN 32000
#define EMBN   512
#define HIDN   1024
#define LATN   128
#define BATCHN 32
#define SEQN   128
#define TDECN  126
#define G4H    (4*HIDN)

// ---------------------------------------------------------------- helpers

__device__ __forceinline__ float sigmoidf_(float x) {
    return 1.0f / (1.0f + __expf(-x));
}
__device__ __forceinline__ float tanhf_(float x) {
    x = fminf(fmaxf(x, -12.0f), 12.0f);
    float e = __expf(2.0f * x);
    return (e - 1.0f) / (e + 1.0f);
}

// A fragment: 16x32 bf16 tile. Lane l: row m = l&15, k-half kg = l>>4.
// elements 0..7  = A[m][k + kg*16 + 0..7]
// elements 8..15 = A[m][k + kg*16 + 8 + 0..7]
__device__ __forceinline__ v16bf load_a_frag(const bhalf* __restrict__ rowk, int koff) {
    v8bf lo = *reinterpret_cast<const v8bf*>(rowk + 2 * koff);
    v8bf hi = *reinterpret_cast<const v8bf*>(rowk + 2 * koff + 8);
    v16bf r;
#pragma unroll
    for (int i = 0; i < 8; ++i) { r[i] = lo[i]; r[i + 8] = hi[i]; }
    return r;
}

__device__ __forceinline__ v8f wmma_bf16(v16bf a, v16bf b, v8f c) {
    return __builtin_amdgcn_wmma_f32_16x16x32_bf16(false, a, false, b, (short)0, c,
                                                   false, false);
}

// ---------------------------------------------------------------- small kernels

__global__ void zero_u32(unsigned int* p, int n) {
    int i = blockIdx.x * blockDim.x + threadIdx.x;
    if (i < n) p[i] = 0u;
}

__global__ void cvt_bf16(const float* __restrict__ in, bhalf* __restrict__ out, int n) {
    int i = blockIdx.x * blockDim.x + threadIdx.x;
    if (i < n) out[i] = (bhalf)in[i];
}

// encoder embedding gather: rows ordered (t*32 + b), row-major EMB
__global__ void embed_enc(const int* __restrict__ tok, const float* __restrict__ emb,
                          bhalf* __restrict__ out) {
    long long tid = (long long)blockIdx.x * blockDim.x + threadIdx.x;
    if (tid >= (long long)BATCHN * SEQN * EMBN) return;
    int e = (int)(tid % EMBN);
    int row = (int)(tid / EMBN);
    int b = row & 31, t = row >> 5;
    int token = tok[b * SEQN + t];
    out[tid] = (bhalf)emb[(long long)token * EMBN + e];
}

// decoder embedding gather: tokens[:,1:-1], rows (t*32 + b), t in [0,126)
__global__ void embed_dec(const int* __restrict__ tok, const float* __restrict__ emb,
                          bhalf* __restrict__ out) {
    long long tid = (long long)blockIdx.x * blockDim.x + threadIdx.x;
    if (tid >= (long long)BATCHN * TDECN * EMBN) return;
    int e = (int)(tid % EMBN);
    int row = (int)(tid / EMBN);
    int b = row & 31, t = row >> 5;
    int token = tok[b * SEQN + t + 1];
    out[tid] = (bhalf)emb[(long long)token * EMBN + e];
}

__global__ void last_idx_kernel(const int* __restrict__ tok, int* __restrict__ lidx) {
    int b = threadIdx.x;
    if (b < BATCHN) {
        int cnt = 0;
        for (int s = 0; s < SEQN; ++s) cnt += (tok[b * SEQN + s] != 0) ? 1 : 0;
        int li = cnt - 1;
        lidx[b] = li < 0 ? 0 : li;
    }
}

__global__ void z_kernel(const float* __restrict__ mean, const float* __restrict__ logvar,
                         const float* __restrict__ eps, bhalf* __restrict__ z) {
    int i = blockIdx.x * blockDim.x + threadIdx.x;
    if (i < BATCHN * LATN)
        z[i] = (bhalf)(mean[i] + eps[i] * __expf(0.5f * logvar[i]));
}

// ---------------------------------------------------------------- WMMA GEMM
// C[M,N] = A[M,K](bf16) * W[N,K](bf16)^T + bias1 + bias2
// AMODE 0: A row r at A + r*lda.
// AMODE 1: logical row r -> (b=r/126, t=r%126), data at A + (t*32+b)*1024 (hd layout).
// Block = 128 threads (4 waves along N). Wave computes MT*16 x NT*16 outputs.
// K-loop is software-pipelined: fragments for k+32 are fetched before the WMMAs of k,
// so the WMMA pipe overlaps with L2/HBM latency (nonzero s_wait_loadcnt).
template <int MT, int NT, int AMODE, int OUTBF>
__global__ void gemm_wmma(const bhalf* __restrict__ A, const bhalf* __restrict__ W,
                          const float* __restrict__ bias1, const float* __restrict__ bias2,
                          void* __restrict__ Cout, int M, int N, int K, int lda) {
    const int lane = threadIdx.x & 31;
    const int wave = threadIdx.x >> 5;
    const int col0 = (blockIdx.x * (blockDim.x >> 5) + wave) * (NT * 16);
    const int row0 = blockIdx.y * (MT * 16);
    if (col0 >= N || row0 >= M) return;  // wave-uniform

    const int koff = (lane >> 4) * 8;

    const bhalf* aptr[MT];
#pragma unroll
    for (int mt = 0; mt < MT; ++mt) {
        int r = row0 + mt * 16 + (lane & 15);
        if (AMODE == 1) {
            int b = r / TDECN;
            int t = r - b * TDECN;
            aptr[mt] = A + (long long)(t * BATCHN + b) * HIDN;
        } else {
            aptr[mt] = A + (long long)r * lda;
        }
    }
    const bhalf* bptr[NT];
#pragma unroll
    for (int nt = 0; nt < NT; ++nt) {
        int c = col0 + nt * 16 + (lane & 15);
        bptr[nt] = W + (long long)c * K + (lane >> 4) * 16;
    }

    v8f acc[MT][NT];
#pragma unroll
    for (int mt = 0; mt < MT; ++mt)
#pragma unroll
        for (int nt = 0; nt < NT; ++nt) acc[mt][nt] = (v8f)(0.0f);

    // prologue: fragments for k = 0
    v16bf af[MT], bfr[NT];
#pragma unroll
    for (int mt = 0; mt < MT; ++mt) af[mt] = load_a_frag(aptr[mt], koff);
#pragma unroll
    for (int nt = 0; nt < NT; ++nt) bfr[nt] = *reinterpret_cast<const v16bf*>(bptr[nt]);

    // steady state: prefetch k+32, compute k
    for (int k = 0; k < K - 32; k += 32) {
        v16bf af_n[MT], bfr_n[NT];
        const int kn = k + 32;
#pragma unroll
        for (int mt = 0; mt < MT; ++mt) af_n[mt] = load_a_frag(aptr[mt] + kn, koff);
#pragma unroll
        for (int nt = 0; nt < NT; ++nt)
            bfr_n[nt] = *reinterpret_cast<const v16bf*>(bptr[nt] + kn);
#pragma unroll
        for (int mt = 0; mt < MT; ++mt)
#pragma unroll
            for (int nt = 0; nt < NT; ++nt)
                acc[mt][nt] = wmma_bf16(af[mt], bfr[nt], acc[mt][nt]);
#pragma unroll
        for (int mt = 0; mt < MT; ++mt) af[mt] = af_n[mt];
#pragma unroll
        for (int nt = 0; nt < NT; ++nt) bfr[nt] = bfr_n[nt];
    }
    // epilogue: last k-slice
#pragma unroll
    for (int mt = 0; mt < MT; ++mt)
#pragma unroll
        for (int nt = 0; nt < NT; ++nt)
            acc[mt][nt] = wmma_bf16(af[mt], bfr[nt], acc[mt][nt]);

#pragma unroll
    for (int nt = 0; nt < NT; ++nt) {
        int col = col0 + nt * 16 + (lane & 15);
        float bv = 0.0f;
        if (bias1) bv += bias1[col];
        if (bias2) bv += bias2[col];
#pragma unroll
        for (int mt = 0; mt < MT; ++mt) {
#pragma unroll
            for (int r = 0; r < 8; ++r) {
                int row = row0 + mt * 16 + r + (lane >> 4) * 8;
                float v = acc[mt][nt][r] + bv;
                if (OUTBF)
                    ((bhalf*)Cout)[(long long)row * N + col] = (bhalf)v;
                else
                    ((float*)Cout)[(long long)row * N + col] = v;
            }
        }
    }
}

// ---------------------------------------------------------------- fused LSTM step
// gates(32,4096) = h_prev(32,1024) @ Whh^T + gpre (gpre already holds x@Wih^T + b).
// Wave wg (0..127): mt = wg>>6 (row tile of 16), jt = wg&63 (hidden col tile of 16).
// Computes all 4 gate tiles at (mt,jt), then the cell update in registers.
// K-loop double-buffered as in gemm_wmma (the scan is latency-critical).
__global__ void lstm_step(const bhalf* __restrict__ h_prev, const float* __restrict__ gpre,
                          const bhalf* __restrict__ Whh, float* __restrict__ c,
                          bhalf* __restrict__ h_out, bhalf* __restrict__ last_h,
                          const int* __restrict__ last_idx, int t) {
    const int lane = threadIdx.x & 31;
    const int wg = blockIdx.x * (blockDim.x >> 5) + (threadIdx.x >> 5);
    const int jt = wg & 63;
    const int mt = wg >> 6;
    const int row0 = mt * 16;
    const int colL = jt * 16 + (lane & 15);
    const int koff = (lane >> 4) * 8;
    const int rsel = (lane >> 4) * 8;

    v8f acc[4];
#pragma unroll
    for (int g = 0; g < 4; ++g)
#pragma unroll
        for (int r = 0; r < 8; ++r)
            acc[g][r] = gpre[(long long)(row0 + r + rsel) * G4H + g * HIDN + colL];

    const bhalf* aptr = h_prev + (long long)(row0 + (lane & 15)) * HIDN;
    const bhalf* bptr[4];
#pragma unroll
    for (int g = 0; g < 4; ++g)
        bptr[g] = Whh + (long long)(g * HIDN + colL) * HIDN + (lane >> 4) * 16;

    // prologue
    v16bf a_c = load_a_frag(aptr, koff);
    v16bf b_c[4];
#pragma unroll
    for (int g = 0; g < 4; ++g) b_c[g] = *reinterpret_cast<const v16bf*>(bptr[g]);

    // steady state: prefetch k+32 fragments, then issue the 4 WMMAs for k
    for (int k = 0; k < HIDN - 32; k += 32) {
        const int kn = k + 32;
        v16bf a_n = load_a_frag(aptr + kn, koff);
        v16bf b_n[4];
#pragma unroll
        for (int g = 0; g < 4; ++g) b_n[g] = *reinterpret_cast<const v16bf*>(bptr[g] + kn);
#pragma unroll
        for (int g = 0; g < 4; ++g) acc[g] = wmma_bf16(a_c, b_c[g], acc[g]);
        a_c = a_n;
#pragma unroll
        for (int g = 0; g < 4; ++g) b_c[g] = b_n[g];
    }
#pragma unroll
    for (int g = 0; g < 4; ++g) acc[g] = wmma_bf16(a_c, b_c[g], acc[g]);

#pragma unroll
    for (int r = 0; r < 8; ++r) {
        int row = row0 + r + rsel;
        long long idx = (long long)row * HIDN + colL;
        float iv = sigmoidf_(acc[0][r]);
        float fv = sigmoidf_(acc[1][r]);
        float gv = tanhf_(acc[2][r]);
        float ov = sigmoidf_(acc[3][r]);
        float cn = fv * c[idx] + iv * gv;
        c[idx] = cn;
        float hn = ov * tanhf_(cn);
        h_out[idx] = (bhalf)hn;
        if (last_idx != nullptr && t == last_idx[row]) last_h[idx] = (bhalf)hn;
    }
}

// ---------------------------------------------------------------- launch

extern "C" void kernel_launch(void* const* d_in, const int* in_sizes, int n_in,
                              void* d_out, int out_size, void* d_ws, size_t ws_size,
                              hipStream_t stream) {
    const int*   tok    = (const int*)d_in[0];
    const float* embE   = (const float*)d_in[1];
    const float* WihE   = (const float*)d_in[2];
    const float* WhhE   = (const float*)d_in[3];
    const float* bihE   = (const float*)d_in[4];
    const float* bhhE   = (const float*)d_in[5];
    const float* Wmu    = (const float*)d_in[6];
    const float* bmu    = (const float*)d_in[7];
    const float* Wlv    = (const float*)d_in[8];
    const float* blv    = (const float*)d_in[9];
    const float* embD   = (const float*)d_in[10];
    const float* Wl2h   = (const float*)d_in[11];
    const float* bl2h   = (const float*)d_in[12];
    const float* WihD   = (const float*)d_in[13];
    const float* WhhD   = (const float*)d_in[14];
    const float* bihD   = (const float*)d_in[15];
    const float* bhhD   = (const float*)d_in[16];
    const float* Wout   = (const float*)d_in[17];
    const float* bout   = (const float*)d_in[18];
    const float* eps    = (const float*)d_in[19];

    float* out = (float*)d_out;
    const long long LOGITS = (long long)BATCHN * TDECN * VOCABN;  // 129024000
    float* out_mean = out + LOGITS;
    float* out_lv   = out_mean + BATCHN * LATN;

    // -------- workspace carve-out (256B aligned)
    char*  ws  = (char*)d_ws;
    size_t off = 0;
    auto alloc = [&](size_t bytes) -> char* {
        char* p = ws + off;
        off = (off + bytes + 255) & ~(size_t)255;
        return p;
    };
    bhalf* WihE_b = (bhalf*)alloc((size_t)G4H * EMBN * 2);
    bhalf* WhhE_b = (bhalf*)alloc((size_t)G4H * HIDN * 2);
    bhalf* WihD_b = (bhalf*)alloc((size_t)G4H * EMBN * 2);
    bhalf* WhhD_b = (bhalf*)alloc((size_t)G4H * HIDN * 2);
    bhalf* Wmu_b  = (bhalf*)alloc((size_t)LATN * HIDN * 2);
    bhalf* Wlv_b  = (bhalf*)alloc((size_t)LATN * HIDN * 2);
    bhalf* Wl2h_b = (bhalf*)alloc((size_t)HIDN * LATN * 2);
    bhalf* Wout_b = (bhalf*)alloc((size_t)VOCABN * HIDN * 2);
    bhalf* Xe     = (bhalf*)alloc((size_t)BATCHN * SEQN * EMBN * 2);
    bhalf* Xd     = (bhalf*)alloc((size_t)BATCHN * TDECN * EMBN * 2);
    float* Gpe    = (float*)alloc((size_t)BATCHN * SEQN * G4H * 4);
    float* Gpd    = (float*)alloc((size_t)BATCHN * TDECN * G4H * 4);
    bhalf* hs     = (bhalf*)alloc((size_t)SEQN * BATCHN * HIDN * 2);
    bhalf* hd     = (bhalf*)alloc((size_t)TDECN * BATCHN * HIDN * 2);
    bhalf* h0b    = (bhalf*)alloc((size_t)BATCHN * HIDN * 2);
    bhalf* lasth  = (bhalf*)alloc((size_t)BATCHN * HIDN * 2);
    bhalf* zb     = (bhalf*)alloc((size_t)BATCHN * LATN * 2);
    float* c_enc  = (float*)alloc((size_t)BATCHN * HIDN * 4);   // contiguous zero region:
    float* c_dec  = (float*)alloc((size_t)BATCHN * HIDN * 4);   //  c_enc, c_dec, hzero
    bhalf* hzero  = (bhalf*)alloc((size_t)BATCHN * HIDN * 2);
    int*   lidx   = (int*)alloc(BATCHN * 4);

    // -------- init: zero c_enc/c_dec/hzero (contiguous), compute last_idx
    {
        int zwords = (BATCHN * HIDN * 4 + BATCHN * HIDN * 4 + BATCHN * HIDN * 2) / 4;
        zero_u32<<<(zwords + 255) / 256, 256, 0, stream>>>((unsigned int*)c_enc, zwords);
        last_idx_kernel<<<1, 32, 0, stream>>>(tok, lidx);
    }

    // -------- weight conversions to bf16
    auto cvt = [&](const float* s, bhalf* d, int n) {
        cvt_bf16<<<(n + 255) / 256, 256, 0, stream>>>(s, d, n);
    };
    cvt(WihE, WihE_b, G4H * EMBN);
    cvt(WhhE, WhhE_b, G4H * HIDN);
    cvt(WihD, WihD_b, G4H * EMBN);
    cvt(WhhD, WhhD_b, G4H * HIDN);
    cvt(Wmu, Wmu_b, LATN * HIDN);
    cvt(Wlv, Wlv_b, LATN * HIDN);
    cvt(Wl2h, Wl2h_b, HIDN * LATN);
    cvt(Wout, Wout_b, VOCABN * HIDN);

    // -------- embedding gathers (bf16, time-major rows t*32+b)
    {
        long long ne = (long long)BATCHN * SEQN * EMBN;
        embed_enc<<<(int)((ne + 255) / 256), 256, 0, stream>>>(tok, embE, Xe);
        long long nd = (long long)BATCHN * TDECN * EMBN;
        embed_dec<<<(int)((nd + 255) / 256), 256, 0, stream>>>(tok, embD, Xd);
    }

    // -------- precompute input-side gates for all timesteps (big parallel GEMMs)
    // Gpe[4096,4096] = Xe @ WihE^T + (bihE + bhhE)
    gemm_wmma<2, 2, 0, 0><<<dim3(G4H / 128, (BATCHN * SEQN) / 32), 128, 0, stream>>>(
        Xe, WihE_b, bihE, bhhE, Gpe, BATCHN * SEQN, G4H, EMBN, EMBN);
    // Gpd[4032,4096] = Xd @ WihD^T + (bihD + bhhD)
    gemm_wmma<2, 2, 0, 0><<<dim3(G4H / 128, (BATCHN * TDECN) / 32), 128, 0, stream>>>(
        Xd, WihD_b, bihD, bhhD, Gpd, BATCHN * TDECN, G4H, EMBN, EMBN);

    // -------- encoder scan
    for (int t = 0; t < SEQN; ++t) {
        const bhalf* hp = (t == 0) ? hzero : hs + (long long)(t - 1) * BATCHN * HIDN;
        lstm_step<<<32, 128, 0, stream>>>(hp, Gpe + (long long)t * BATCHN * G4H, WhhE_b,
                                          c_enc, hs + (long long)t * BATCHN * HIDN,
                                          lasth, lidx, t);
    }

    // -------- mean / log_var (write straight into d_out)
    gemm_wmma<2, 1, 0, 0><<<dim3(LATN / 64, 1), 128, 0, stream>>>(
        lasth, Wmu_b, bmu, nullptr, out_mean, BATCHN, LATN, HIDN, HIDN);
    gemm_wmma<2, 1, 0, 0><<<dim3(LATN / 64, 1), 128, 0, stream>>>(
        lasth, Wlv_b, blv, nullptr, out_lv, BATCHN, LATN, HIDN, HIDN);

    // -------- reparameterize, z -> h0 (bf16 out)
    z_kernel<<<(BATCHN * LATN + 255) / 256, 256, 0, stream>>>(out_mean, out_lv, eps, zb);
    gemm_wmma<2, 2, 0, 1><<<dim3(HIDN / 128, 1), 128, 0, stream>>>(
        zb, Wl2h_b, bl2h, nullptr, h0b, BATCHN, HIDN, LATN, LATN);

    // -------- decoder scan (h stored directly into hd, consumed by output GEMM)
    for (int t = 0; t < TDECN; ++t) {
        const bhalf* hp = (t == 0) ? h0b : hd + (long long)(t - 1) * BATCHN * HIDN;
        lstm_step<<<32, 128, 0, stream>>>(hp, Gpd + (long long)t * BATCHN * G4H, WhhD_b,
                                          c_dec, hd + (long long)t * BATCHN * HIDN,
                                          nullptr, nullptr, t);
    }

    // -------- output projection: logits[4032,32000] = hd @ Wout^T + bout
    // AMODE=1 remaps logical row (b*126+t) -> hd's time-major storage.
    gemm_wmma<2, 2, 1, 0><<<dim3(VOCABN / 128, (BATCHN * TDECN) / 32), 128, 0, stream>>>(
        hd, Wout_b, bout, nullptr, out, BATCHN * TDECN, VOCABN, HIDN, HIDN);

    (void)in_sizes; (void)n_in; (void)out_size; (void)ws_size;
}
